// CSWinBlock_2723009266306
// MI455X (gfx1250) — matrix operand order repaired and checked by reading, not compiled, not run on hardware
//
#include <hip/hip_runtime.h>
#include <hip/hip_bf16.h>
#include <cmath>

// ---------------------------------------------------------------------------
// CSWin block on gfx1250: all matmuls via v_wmma_f32_16x16x32_f16 (f16 in,
// f32 accumulate).  LN / softmax statistics / conv / epilogues in fp32.
// ---------------------------------------------------------------------------

typedef __attribute__((ext_vector_type(16))) _Float16 v16h;
typedef __attribute__((ext_vector_type(8)))  _Float16 v8h;
typedef __attribute__((ext_vector_type(8)))  float    v8f;

#define RESO   112
#define LTOK   (RESO * RESO)          // 12544
#define BATCH  2
#define MROWS  (BATCH * LTOK)         // 25088
#define CDIM   128
#define HIDD   512
#define QKVC   384

__device__ __forceinline__ v8f wmma16(v16h a, v16h b, v8f c) {
  // D = A(16x32 f16) * B(32x16 f16) + C(16x16 f32)
  return __builtin_amdgcn_wmma_f32_16x16x32_f16(
      /*neg_a=*/false, a, /*neg_b=*/false, b,
      /*c_mod=*/(short)0, c, /*reuse_a=*/false, /*reuse_b=*/false);
}

__device__ __forceinline__ v8f vzero8() {
  v8f z;
#pragma unroll
  for (int r = 0; r < 8; ++r) z[r] = 0.f;
  return z;
}

// window-local index l -> global token within one image
template <int BR>
__device__ __forceinline__ int win_token(int w, int l) {
  if (BR == 0) {                       // 112 x 7 vertical stripe, window = cols [7w,7w+7)
    int i = l / 7, j = l - i * 7;
    return i * RESO + w * 7 + j;
  } else {                             // 7 x 112 horizontal stripe, window = rows [7w,7w+7)
    int i = l / RESO, j = l - i * RESO;
    return (w * 7 + i) * RESO + j;
  }
}

// ---------------------------------------------------------------------------
// fp32 -> fp16 weight conversion
// ---------------------------------------------------------------------------
__global__ void cvt_f16(const float* __restrict__ s, _Float16* __restrict__ d, int n) {
  int i = blockIdx.x * 256 + threadIdx.x;
  if (i < n) d[i] = (_Float16)s[i];
}

// ---------------------------------------------------------------------------
// LayerNorm (fp32 in -> fp16 out), one wave per 128-channel row
// ---------------------------------------------------------------------------
__global__ __launch_bounds__(256)
void ln_f16(const float* __restrict__ x, const float* __restrict__ g,
            const float* __restrict__ b, _Float16* __restrict__ out, int M) {
  const int wave = threadIdx.x >> 5, lane = threadIdx.x & 31;
  const int row = blockIdx.x * 8 + wave;
  if (row >= M) return;
  const float* xr = x + (size_t)row * CDIM;
  float v[4], s = 0.f;
#pragma unroll
  for (int i = 0; i < 4; ++i) { v[i] = xr[lane + 32 * i]; s += v[i]; }
#pragma unroll
  for (int off = 1; off < 32; off <<= 1) s += __shfl_xor(s, off, 32);
  const float mean = s * (1.f / CDIM);
  float q = 0.f;
#pragma unroll
  for (int i = 0; i < 4; ++i) { float d = v[i] - mean; q += d * d; }
#pragma unroll
  for (int off = 1; off < 32; off <<= 1) q += __shfl_xor(q, off, 32);
  const float inv = rsqrtf(q * (1.f / CDIM) + 1e-5f);
  _Float16* orow = out + (size_t)row * CDIM;
#pragma unroll
  for (int i = 0; i < 4; ++i) {
    const int c = lane + 32 * i;
    orow[c] = (_Float16)((v[i] - mean) * inv * g[c] + b[c]);
  }
}

// ---------------------------------------------------------------------------
// Generic WMMA GEMM:  out[M,N] = A[M,K] * W[N,K]^T (+ epilogue)
//   EPI 0: store f16          EPI 1: + bias + resid, store f32
//   EPI 2: + bias, exact GELU, store f16
// Block = 8 waves, tile 128(M) x 64(N); wave tile 16 x 64.
// ---------------------------------------------------------------------------
template <int EPI, bool AF32>
__global__ __launch_bounds__(256, 2)
void gemm_wmma(const void* __restrict__ Ap, const _Float16* __restrict__ W,
               const float* __restrict__ bias, const float* __restrict__ resid,
               void* __restrict__ out, int M, int N, int K) {
  const int wave = threadIdx.x >> 5;
  const int lane = threadIdx.x & 31;
  const int m = lane & 15;
  const int grp = lane >> 4;
  const int row0 = blockIdx.y * 128 + wave * 16;
  const int col0 = blockIdx.x * 64;
  (void)M;

  v8f acc[4];
#pragma unroll
  for (int i = 0; i < 4; ++i) acc[i] = vzero8();

  for (int k0 = 0; k0 < K; k0 += 32) {
    // A fragment (16x32, ISA 16-bit A layout)
    v16h a;
    const int kh = grp * 8;
    if constexpr (AF32) {
      const float* Af = (const float*)Ap + (size_t)(row0 + m) * K + k0;
#pragma unroll
      for (int e = 0; e < 8; ++e) {
        a[e]     = (_Float16)Af[kh + e];
        a[8 + e] = (_Float16)Af[16 + kh + e];
      }
    } else {
      const _Float16* Ah = (const _Float16*)Ap + (size_t)(row0 + m) * K + k0;
      v8h lo = *(const v8h*)(Ah + kh);
      v8h hi = *(const v8h*)(Ah + 16 + kh);
#pragma unroll
      for (int e = 0; e < 8; ++e) { a[e] = lo[e]; a[8 + e] = hi[e]; }
    }
#pragma unroll
    for (int nt = 0; nt < 4; ++nt) {
      // B fragment: B[k,n] = W[n,k]; two 16B halves per lane (k = 8*grp.., 16+8*grp..).
      const _Float16* Wp = W + (size_t)(col0 + nt * 16 + m) * K + k0 + grp * 8;
      v8h blo = *(const v8h*)Wp;
      v8h bhi = *(const v8h*)(Wp + 16);
      v16h b;
#pragma unroll
      for (int e = 0; e < 8; ++e) { b[e] = blo[e]; b[8 + e] = bhi[e]; }
      acc[nt] = wmma16(a, b, acc[nt]);
    }
  }

#pragma unroll
  for (int nt = 0; nt < 4; ++nt) {
#pragma unroll
    for (int r = 0; r < 8; ++r) {
      const int row = row0 + grp * 8 + r;
      const int col = col0 + nt * 16 + m;
      const size_t o = (size_t)row * N + col;
      float v = acc[nt][r];
      if constexpr (EPI == 0) {
        ((_Float16*)out)[o] = (_Float16)v;
      } else if constexpr (EPI == 1) {
        ((float*)out)[o] = v + bias[col] + resid[o];
      } else {
        v += bias[col];
        ((_Float16*)out)[o] =
            (_Float16)(0.5f * v * (1.f + erff(v * 0.70710678118654752f)));
      }
    }
  }
}

// ---------------------------------------------------------------------------
// Flash-style window attention. One block per (window,head) x 4 q-tiles.
// seq = 784 tokens/window, hd = 32 (exactly one WMMA K step).
// qkv layout: [B*L, 384] f16 (q:0..127, k:128..255, v:256..383).
// Output: fp32 att[B*L, 128] (before LePE).
// ---------------------------------------------------------------------------
template <int BR>
__global__ __launch_bounds__(128, 4)
void attn_wmma(const _Float16* __restrict__ qkv, float* __restrict__ att) {
  __shared__ int tokLUT[784];
  __shared__ _Float16 pT[4][16 * 36];   // per-wave P tile, padded stride

  const int unit = blockIdx.y;          // 64 units: b(2) * window(16) * head(2)
  const int bI = unit >> 5;
  const int rem = unit & 31;
  const int w = rem >> 1;
  const int h = rem & 1;

  for (int i = threadIdx.x; i < 784; i += 128) tokLUT[i] = win_token<BR>(w, i);
  __syncthreads();

  const int wave = threadIdx.x >> 5;
  const int lane = threadIdx.x & 31;
  const int qt = blockIdx.x * 4 + wave;
  if (qt >= 49) return;                 // whole-wave exit: EXEC stays all-ones

  const int m = lane & 15, grp = lane >> 4;
  const int ch0 = BR * 64 + h * 32;
  const size_t base = (size_t)bI * LTOK;
  const float SCALE = 0.17677669529663687f;  // 32^-0.5

  // Q fragment (scaled), rows = q-tile tokens
  v16h qa;
  {
    const int tok = tokLUT[qt * 16 + m];
    const _Float16* qp = qkv + (base + tok) * QKVC + ch0;
    const int kh = grp * 16;
#pragma unroll
    for (int e = 0; e < 8; ++e) {
      qa[e]     = (_Float16)((float)qp[kh + e] * SCALE);
      qa[8 + e] = (_Float16)((float)qp[8 + kh + e] * SCALE);
    }
  }

  v8f acc0 = vzero8(), acc1 = vzero8();
  float mS[8], lS[8];
#pragma unroll
  for (int r = 0; r < 8; ++r) { mS[r] = -1e30f; lS[r] = 0.f; }

  for (int kc = 0; kc < 25; ++kc) {     // 784 keys = 24*32 + 16
    const int kb0 = kc * 32;
    v8f s0, s1;
    {
      int key = kb0 + m; if (key > 783) key = 783;
      const _Float16* kp =
          qkv + (base + tokLUT[key]) * QKVC + 128 + ch0 + grp * 16;
      s0 = wmma16(qa, *(const v16h*)kp, vzero8());
    }
    {
      int key = kb0 + 16 + m; if (key > 783) key = 783;
      const _Float16* kp =
          qkv + (base + tokLUT[key]) * QKVC + 128 + ch0 + grp * 16;
      s1 = wmma16(qa, *(const v16h*)kp, vzero8());
    }
    if (kb0 + 16 >= 784) {              // mask invalid tail keys
#pragma unroll
      for (int r = 0; r < 8; ++r) s1[r] = -1e30f;
    }

    // online softmax (row = r + 8*grp, reduce across the 16 lanes per group)
    float mx[8];
#pragma unroll
    for (int r = 0; r < 8; ++r) mx[r] = fmaxf(s0[r], s1[r]);
#pragma unroll
    for (int off = 1; off < 16; off <<= 1)
#pragma unroll
      for (int r = 0; r < 8; ++r) mx[r] = fmaxf(mx[r], __shfl_xor(mx[r], off, 32));

    float p0[8], p1[8], sc[8], rs[8];
#pragma unroll
    for (int r = 0; r < 8; ++r) {
      const float mn = fmaxf(mS[r], mx[r]);
      sc[r] = __expf(mS[r] - mn);
      p0[r] = __expf(s0[r] - mn);
      p1[r] = __expf(s1[r] - mn);
      mS[r] = mn;
      rs[r] = p0[r] + p1[r];
    }
#pragma unroll
    for (int off = 1; off < 16; off <<= 1)
#pragma unroll
      for (int r = 0; r < 8; ++r) rs[r] += __shfl_xor(rs[r], off, 32);
#pragma unroll
    for (int r = 0; r < 8; ++r) {
      lS[r] = lS[r] * sc[r] + rs[r];
      acc0[r] *= sc[r];
      acc1[r] *= sc[r];
    }

    // P (C layout) -> LDS -> A fragment layout
    _Float16* pw = pT[wave];
#pragma unroll
    for (int r = 0; r < 8; ++r) {
      const int rr = grp * 8 + r;
      pw[rr * 36 + m]      = (_Float16)p0[r];
      pw[rr * 36 + 16 + m] = (_Float16)p1[r];
    }
    v16h pa;
    {
      const int kh = grp * 8;
#pragma unroll
      for (int e = 0; e < 8; ++e) {
        pa[e]     = pw[m * 36 + kh + e];
        pa[8 + e] = pw[m * 36 + 16 + kh + e];
      }
    }

    // PV: V fragment rows gathered per key
#pragma unroll
    for (int nt = 0; nt < 2; ++nt) {
      v16h vf;
#pragma unroll
      for (int e = 0; e < 16; ++e) {
        int key = kb0 + ((e & 8) ? 16 : 0) + grp * 8 + (e & 7); if (key > 783) key = 783;
        vf[e] = qkv[(base + tokLUT[key]) * QKVC + 256 + ch0 + nt * 16 + m];
      }
      if (nt == 0) acc0 = wmma16(pa, vf, acc0);
      else         acc1 = wmma16(pa, vf, acc1);
    }
  }

  // normalize + scatter back to token layout
#pragma unroll
  for (int r = 0; r < 8; ++r) {
    const int rr = grp * 8 + r;
    const int tok = tokLUT[qt * 16 + rr];
    const float inv = 1.f / lS[r];
    float* op = att + (base + tok) * CDIM + ch0;
    op[m]      = acc0[r] * inv;
    op[16 + m] = acc1[r] * inv;
  }
}

// ---------------------------------------------------------------------------
// LePE: depthwise 3x3 conv of V inside each stripe window, added to att.
// ---------------------------------------------------------------------------
__global__ __launch_bounds__(256)
void lepe_add(const _Float16* __restrict__ qkv,
              const float* __restrict__ cw0, const float* __restrict__ cb0,
              const float* __restrict__ cw1, const float* __restrict__ cb1,
              float* __restrict__ att) {
  const size_t idx = (size_t)blockIdx.x * 256 + threadIdx.x;
  if (idx >= (size_t)MROWS * CDIM) return;
  const int ch = (int)(idx & (CDIM - 1));
  const size_t tg = idx >> 7;           // b*L + t
  const int t = (int)(tg % LTOK);
  const int bI = (int)(tg / LTOK);
  const int row = t / RESO, col = t % RESO;

  const bool br1 = (ch >= 64);
  const int cc = br1 ? ch - 64 : ch;
  const float* cw = (br1 ? cw1 : cw0) + cc * 9;
  float acc = (br1 ? cb1 : cb0)[cc];
  const int wseg = br1 ? (row / 7) : (col / 7);

#pragma unroll
  for (int dy = -1; dy <= 1; ++dy)
#pragma unroll
    for (int dx = -1; dx <= 1; ++dx) {
      const int r2 = row + dy, c2 = col + dx;
      bool ok = (r2 >= 0) && (r2 < RESO) && (c2 >= 0) && (c2 < RESO);
      if (ok) ok = br1 ? (r2 / 7 == wseg) : (c2 / 7 == wseg);
      if (ok) {
        const size_t vi =
            ((size_t)bI * LTOK + (size_t)r2 * RESO + c2) * QKVC + 256 + ch;
        acc += cw[(dy + 1) * 3 + (dx + 1)] * (float)qkv[vi];
      }
    }
  att[idx] += acc;
}

// ---------------------------------------------------------------------------
// Launch sequence
// ---------------------------------------------------------------------------
extern "C" void kernel_launch(void* const* d_in, const int* in_sizes, int n_in,
                              void* d_out, int out_size, void* d_ws, size_t ws_size,
                              hipStream_t stream) {
  (void)in_sizes; (void)n_in; (void)out_size; (void)ws_size;
  const float* x      = (const float*)d_in[0];
  const float* g1     = (const float*)d_in[1];
  const float* b1     = (const float*)d_in[2];
  const float* w_qkv  = (const float*)d_in[3];
  const float* cw0    = (const float*)d_in[4];
  const float* cb0    = (const float*)d_in[5];
  const float* cw1    = (const float*)d_in[6];
  const float* cb1    = (const float*)d_in[7];
  const float* w_proj = (const float*)d_in[8];
  const float* b_proj = (const float*)d_in[9];
  const float* g2     = (const float*)d_in[10];
  const float* b2     = (const float*)d_in[11];
  const float* w_fc1  = (const float*)d_in[12];
  const float* b_fc1  = (const float*)d_in[13];
  const float* w_fc2  = (const float*)d_in[14];
  const float* b_fc2  = (const float*)d_in[15];

  char* ws = (char*)d_ws;
  // fp16 weights: 384*128 + 128*128 + 512*128 + 128*512 = 196608 halfs (384 KB)
  _Float16* wbq = (_Float16*)ws;
  _Float16* wbp = wbq + 384 * 128;
  _Float16* wb1 = wbp + 128 * 128;
  _Float16* wb2 = wb1 + 512 * 128;
  char* pA = ws + 393216;                        // region A: qkv f16 / later h f16
  _Float16* qkvb = (_Float16*)pA;
  _Float16* hbuf = (_Float16*)pA;                // reuse (qkv dead after LePE)
  char* pB = pA + (size_t)MROWS * HIDD * 2;      // att f32
  float* att = (float*)pB;
  char* pC = pB + (size_t)MROWS * CDIM * 4;      // x after proj+resid, f32
  float* xb = (float*)pC;
  char* pD = pC + (size_t)MROWS * CDIM * 4;      // LN output f16 (reused)
  _Float16* lnb = (_Float16*)pD;

  // 1) convert weights to fp16
  cvt_f16<<<(384 * 128 + 255) / 256, 256, 0, stream>>>(w_qkv, wbq, 384 * 128);
  cvt_f16<<<(128 * 128 + 255) / 256, 256, 0, stream>>>(w_proj, wbp, 128 * 128);
  cvt_f16<<<(512 * 128 + 255) / 256, 256, 0, stream>>>(w_fc1, wb1, 512 * 128);
  cvt_f16<<<(128 * 512 + 255) / 256, 256, 0, stream>>>(w_fc2, wb2, 128 * 512);

  // 2) LN1 -> fp16
  ln_f16<<<MROWS / 8, 256, 0, stream>>>(x, g1, b1, lnb, MROWS);

  // 3) QKV GEMM (no bias) -> fp16 qkv
  gemm_wmma<0, false><<<dim3(QKVC / 64, MROWS / 128), 256, 0, stream>>>(
      lnb, wbq, nullptr, nullptr, qkvb, MROWS, QKVC, CDIM);

  // 4) window attention, both stripe orientations
  attn_wmma<0><<<dim3(13, 64), 128, 0, stream>>>(qkvb, att);
  attn_wmma<1><<<dim3(13, 64), 128, 0, stream>>>(qkvb, att);

  // 5) LePE depthwise conv added to att
  lepe_add<<<(MROWS * CDIM) / 256, 256, 0, stream>>>(qkvb, cw0, cb0, cw1, cb1, att);

  // 6) proj GEMM + bias + residual(x) -> xb (f32)
  gemm_wmma<1, true><<<dim3(CDIM / 64, MROWS / 128), 256, 0, stream>>>(
      att, wbp, b_proj, x, xb, MROWS, CDIM, CDIM);

  // 7) LN2 -> fp16
  ln_f16<<<MROWS / 8, 256, 0, stream>>>(xb, g2, b2, lnb, MROWS);

  // 8) FC1 + bias + exact GELU -> fp16 h (overlaps dead qkv region)
  gemm_wmma<2, false><<<dim3(HIDD / 64, MROWS / 128), 256, 0, stream>>>(
      lnb, wb1, b_fc1, nullptr, hbuf, MROWS, HIDD, CDIM);

  // 9) FC2 + bias + residual(xb) -> d_out (f32)
  gemm_wmma<1, false><<<dim3(CDIM / 64, MROWS / 128), 256, 0, stream>>>(
      hbuf, wb2, b_fc2, xb, (float*)d_out, MROWS, CDIM, HIDD);
}
